// SwinJSCC_Decoder_68521908240731
// MI455X (gfx1250) — matrix operand order repaired and checked by reading, not compiled, not run on hardware
//
#include <hip/hip_runtime.h>

typedef __attribute__((ext_vector_type(16))) _Float16 v16h;
typedef __attribute__((ext_vector_type(8)))  _Float16 v8h;
typedef __attribute__((ext_vector_type(8)))  float    v8f;

// ---------------------------------------------------------------------------
// WMMA 16x16x32 f16 tile: A row-major (M x K), BT n-major (N x K) so that both
// fragments are contiguous v8h (ds_load_b128) chunks per ISA 7.12.2:
//   A frag lane(r,half): halves [r*lda + half*8 .. +7] and [r*lda+16+half*8 .. +7]
//   B frag lane(c,half): halves [c*ldbt + half*8 .. +7] and [c*ldbt+16+half*8 .. +7]
// ---------------------------------------------------------------------------
__device__ __forceinline__ v8f wmma_tile(const _Float16* A, int lda,
                                         const _Float16* BT, int ldbt, v8f acc) {
    const int lane = threadIdx.x & 31;
    const int half = lane >> 4;
    const int r    = lane & 15;
    const v8h a0 = *(const v8h*)(A + r * lda + (half << 3));
    const v8h a1 = *(const v8h*)(A + r * lda + 16 + (half << 3));
    const v8h b0 = *(const v8h*)(BT + r * ldbt + (half << 3));
    const v8h b1 = *(const v8h*)(BT + r * ldbt + 16 + (half << 3));
    v16h a, b;
#pragma unroll
    for (int j = 0; j < 8; ++j) {
        a[j] = a0[j]; a[j + 8] = a1[j];
        b[j] = b0[j]; b[j + 8] = b1[j];
    }
    return __builtin_amdgcn_wmma_f32_16x16x32_f16(false, a, false, b,
                                                  (short)0, acc, false, false);
}

// ---------------------------------------------------------------------------
// Generic GEMM: C[M,N] = epilogue(A[M,K] @ W[K,N])
// REQUIRES: M % 64 == 0, K % 32 == 0 (true for every call in this network).
// N may have a tail (handled by a uniform per-block guarded path).
//   act: 0 = none, 1 = +bias, 2 = +bias,GELU(exact), 3 = +bias,sigmoid
//   residual != nullptr -> added at the end (fused residual)
// Block = 128 threads = 4 waves; block tile 64x32; each wave: 16x32 strip
// (A fragment reused across two WMMAs). B tile staged transposed in LDS.
// ---------------------------------------------------------------------------
__global__ void __launch_bounds__(128)
gemm_wmma_kernel(const float* __restrict__ A, const float* __restrict__ W,
                 const float* __restrict__ bias, const float* __restrict__ residual,
                 float* __restrict__ C, int M, int N, int K, int act) {
    __shared__ __attribute__((aligned(16))) _Float16 As[64 * 32];   // row-major
    __shared__ __attribute__((aligned(16))) _Float16 BsT[32 * 32];  // n-major

    const int tid  = threadIdx.x;
    const int wave = tid >> 5;
    const int lane = tid & 31;
    const int m0 = blockIdx.y << 6;
    const int n0 = blockIdx.x << 5;
    const int wm = wave << 4;            // wave's 16-row strip within the 64

    const bool ntail = (n0 + 32) > N;    // uniform across the block

    v8f acc0 = {0.f, 0.f, 0.f, 0.f, 0.f, 0.f, 0.f, 0.f};
    v8f acc1 = acc0;

    // staging thread maps (fixed across K loop)
    const int ar = tid >> 1;             // 0..63 : A row
    const int ac = (tid & 1) << 4;       // 0|16  : A col base (16 floats)
    const int bk = tid >> 2;             // 0..31 : B k-row
    const int bn = (tid & 3) << 3;       // 0|8|16|24 : B n base (8 floats)

    for (int k0 = 0; k0 < K; k0 += 32) {
        // ---- stage A 64x32 (vector b128 loads, no guards needed) ----
        {
            const float* src = A + (size_t)(m0 + ar) * K + k0 + ac;
            float4 f0 = *(const float4*)(src);
            float4 f1 = *(const float4*)(src + 4);
            float4 f2 = *(const float4*)(src + 8);
            float4 f3 = *(const float4*)(src + 12);
            v8h h0, h1;
            h0[0] = (_Float16)f0.x; h0[1] = (_Float16)f0.y; h0[2] = (_Float16)f0.z; h0[3] = (_Float16)f0.w;
            h0[4] = (_Float16)f1.x; h0[5] = (_Float16)f1.y; h0[6] = (_Float16)f1.z; h0[7] = (_Float16)f1.w;
            h1[0] = (_Float16)f2.x; h1[1] = (_Float16)f2.y; h1[2] = (_Float16)f2.z; h1[3] = (_Float16)f2.w;
            h1[4] = (_Float16)f3.x; h1[5] = (_Float16)f3.y; h1[6] = (_Float16)f3.z; h1[7] = (_Float16)f3.w;
            *(v8h*)&As[ar * 32 + ac]     = h0;
            *(v8h*)&As[ar * 32 + ac + 8] = h1;
        }
        // ---- stage B 32x32 transposed into BsT[n][k] ----
        if (!ntail) {
            const float* src = W + (size_t)(k0 + bk) * N + n0 + bn;
            float4 f0 = *(const float4*)(src);
            float4 f1 = *(const float4*)(src + 4);
            BsT[(bn + 0) * 32 + bk] = (_Float16)f0.x;
            BsT[(bn + 1) * 32 + bk] = (_Float16)f0.y;
            BsT[(bn + 2) * 32 + bk] = (_Float16)f0.z;
            BsT[(bn + 3) * 32 + bk] = (_Float16)f0.w;
            BsT[(bn + 4) * 32 + bk] = (_Float16)f1.x;
            BsT[(bn + 5) * 32 + bk] = (_Float16)f1.y;
            BsT[(bn + 6) * 32 + bk] = (_Float16)f1.z;
            BsT[(bn + 7) * 32 + bk] = (_Float16)f1.w;
        } else {
#pragma unroll
            for (int j = 0; j < 8; ++j) {
                int gn = n0 + bn + j;
                float v = (gn < N) ? W[(size_t)(k0 + bk) * N + gn] : 0.f;
                BsT[(bn + j) * 32 + bk] = (_Float16)v;
            }
        }
        __syncthreads();

        // prefetch next K tile while WMMA runs
        if (k0 + 32 < K) {
            __builtin_prefetch(A + (size_t)(m0 + ar) * K + k0 + 32 + ac, 0, 1);
            if (!ntail)
                __builtin_prefetch(W + (size_t)(k0 + 32 + bk) * N + n0 + bn, 0, 1);
        }

        acc0 = wmma_tile(&As[wm * 32], 32, &BsT[0], 32, acc0);
        acc1 = wmma_tile(&As[wm * 32], 32, &BsT[16 * 32], 32, acc1);
        __syncthreads();
    }

    // ---- epilogue ----
    const int col   = lane & 15;
    const int rbase = (lane >> 4) << 3;
    v8f accs[2] = {acc0, acc1};
#pragma unroll
    for (int t = 0; t < 2; ++t) {
        const int gn = n0 + (t << 4) + col;
        if (gn < N) {
            const float bv = (act >= 1 && bias) ? bias[gn] : 0.f;
#pragma unroll
            for (int v = 0; v < 8; ++v) {
                int gm = m0 + wm + rbase + v;
                if (gm < M) {
                    float val = accs[t][v] + bv;
                    if (act == 2)      val = 0.5f * val * (1.0f + erff(val * 0.7071067811865475f));
                    else if (act == 3) val = 1.0f / (1.0f + __expf(-val));
                    if (residual)      val += residual[(size_t)gm * N + gn];
                    C[(size_t)gm * N + gn] = val;
                }
            }
        }
    }
}

// ---------------------------------------------------------------------------
// Fused window attention: one block (256 thr = 8 waves) per (window, head).
// head_dim == 32 in every stage, window = 64 tokens.
// S = QK^T (4x4 WMMA tiles; K stored row-major == BT of Q.K^T), scale +
// on-the-fly rel-pos bias + analytic shift mask -> softmax in LDS ->
// O = P.V (V stored transposed so B fragments are contiguous).
// ---------------------------------------------------------------------------
__global__ void __launch_bounds__(256)
attn_kernel(const float* __restrict__ qkv, const float* __restrict__ rpb,
            float* __restrict__ out, int dim, int nh, int H, int W, int shifted) {
    __shared__ __attribute__((aligned(16))) _Float16 hbuf[2 * 64 * 32]; // Q | K, later P
    __shared__ __attribute__((aligned(16))) _Float16 hVT[32 * 64];      // V transposed [d][t]
    __shared__ float S[64 * 64];

    const int win  = blockIdx.x;
    const int h    = blockIdx.y;
    const int nWx  = W >> 3;
    const int nW   = (H >> 3) * nWx;
    const int widx = win % nW;
    const int wh   = widx / nWx;
    const int ww   = widx % nWx;
    const int tid  = threadIdx.x;
    const int wave = tid >> 5;
    const int lane = tid & 31;

    _Float16* hQ = hbuf;
    _Float16* hK = hbuf + 64 * 32;

    // ---- vectorized QKV load: thread -> (token t, 8 channels at d0) ----
    {
        const int t  = tid >> 2;
        const int d0 = (tid & 3) << 3;
        const size_t base = (size_t)win * 64 * (3 * dim);
        const float* src = qkv + base + (size_t)t * (3 * dim) + h * 32 + d0;
        float4 q0 = *(const float4*)(src);
        float4 q1 = *(const float4*)(src + 4);
        float4 k0 = *(const float4*)(src + dim);
        float4 k1 = *(const float4*)(src + dim + 4);
        float4 v0 = *(const float4*)(src + 2 * dim);
        float4 v1 = *(const float4*)(src + 2 * dim + 4);
        v8h hq, hk;
        hq[0] = (_Float16)q0.x; hq[1] = (_Float16)q0.y; hq[2] = (_Float16)q0.z; hq[3] = (_Float16)q0.w;
        hq[4] = (_Float16)q1.x; hq[5] = (_Float16)q1.y; hq[6] = (_Float16)q1.z; hq[7] = (_Float16)q1.w;
        hk[0] = (_Float16)k0.x; hk[1] = (_Float16)k0.y; hk[2] = (_Float16)k0.z; hk[3] = (_Float16)k0.w;
        hk[4] = (_Float16)k1.x; hk[5] = (_Float16)k1.y; hk[6] = (_Float16)k1.z; hk[7] = (_Float16)k1.w;
        *(v8h*)&hQ[t * 32 + d0] = hq;
        *(v8h*)&hK[t * 32 + d0] = hk;
        hVT[(d0 + 0) * 64 + t] = (_Float16)v0.x;
        hVT[(d0 + 1) * 64 + t] = (_Float16)v0.y;
        hVT[(d0 + 2) * 64 + t] = (_Float16)v0.z;
        hVT[(d0 + 3) * 64 + t] = (_Float16)v0.w;
        hVT[(d0 + 4) * 64 + t] = (_Float16)v1.x;
        hVT[(d0 + 5) * 64 + t] = (_Float16)v1.y;
        hVT[(d0 + 6) * 64 + t] = (_Float16)v1.z;
        hVT[(d0 + 7) * 64 + t] = (_Float16)v1.w;
    }
    __syncthreads();

    const float scale = 0.17677669529663687f;  // 32^-0.5

    // ---- S = Q @ K^T : 16 tiles, wave handles 2 (uniform loop)
    for (int ti = wave; ti < 16; ti += 8) {
        int tm = ti >> 2, tn = ti & 3;
        v8f acc = {0.f, 0.f, 0.f, 0.f, 0.f, 0.f, 0.f, 0.f};
        acc = wmma_tile(&hQ[tm * 16 * 32], 32, &hK[tn * 16 * 32], 32, acc);

        const int col   = lane & 15;
        const int rbase = (lane >> 4) << 3;
        const int kk = tn * 16 + col;
        const int ky = kk >> 3, kx = kk & 7;
#pragma unroll
        for (int v = 0; v < 8; ++v) {
            int q  = tm * 16 + rbase + v;
            int qy = q >> 3, qx = q & 7;
            int ridx = (qy - ky + 7) * 15 + (qx - kx + 7);
            float val = acc[v] * scale + rpb[ridx * nh + h];
            if (shifted) {
                int gqy = wh * 8 + qy, gqx = ww * 8 + qx;
                int gky = wh * 8 + ky, gkx = ww * 8 + kx;
                int rq = (gqy < H - 8 ? 0 : (gqy < H - 4 ? 1 : 2)) * 3 +
                         (gqx < W - 8 ? 0 : (gqx < W - 4 ? 1 : 2));
                int rk = (gky < H - 8 ? 0 : (gky < H - 4 ? 1 : 2)) * 3 +
                         (gkx < W - 8 ? 0 : (gkx < W - 4 ? 1 : 2));
                if (rq != rk) val += -100.0f;
            }
            S[q * 64 + kk] = val;
        }
    }
    __syncthreads();

    // ---- row softmax (64 rows)
    if (tid < 64) {
        float* row = &S[tid * 64];
        float m = row[0];
        for (int j = 1; j < 64; ++j) m = fmaxf(m, row[j]);
        float s = 0.f;
        for (int j = 0; j < 64; ++j) { float e = __expf(row[j] - m); row[j] = e; s += e; }
        float inv = 1.0f / s;
        for (int j = 0; j < 64; ++j) row[j] *= inv;
    }
    __syncthreads();

    // ---- P (f16, row-major 64x64) overwrites Q|K region
    for (int i = tid; i < 64 * 64; i += 256) hbuf[i] = (_Float16)S[i];
    __syncthreads();

    // ---- O = P @ V : 8 tiles (4x2), one per wave; K loop {0,32}
    {
        const int tm = wave >> 1, tn = wave & 1;
        v8f acc = {0.f, 0.f, 0.f, 0.f, 0.f, 0.f, 0.f, 0.f};
        acc = wmma_tile(&hbuf[tm * 16 * 64],      64, &hVT[tn * 16 * 64],      64, acc);
        acc = wmma_tile(&hbuf[tm * 16 * 64 + 32], 64, &hVT[tn * 16 * 64 + 32], 64, acc);
        const int col   = lane & 15;
        const int rbase = (lane >> 4) << 3;
#pragma unroll
        for (int v = 0; v < 8; ++v) {
            int t = tm * 16 + rbase + v;
            int d = tn * 16 + col;
            out[((size_t)win * 64 + t) * dim + h * 32 + d] = acc[v];
        }
    }
}

// ---------------------------------------------------------------------------
// LayerNorm: one wave per row
// ---------------------------------------------------------------------------
__global__ void ln_kernel(const float* __restrict__ x, const float* __restrict__ g,
                          const float* __restrict__ b, float* __restrict__ y, int C) {
    const size_t row = blockIdx.x;
    const float* xr = x + row * C;
    float s = 0.f, s2 = 0.f;
    for (int i = threadIdx.x; i < C; i += 32) { float v = xr[i]; s += v; s2 += v * v; }
    for (int o = 16; o > 0; o >>= 1) {
        s  += __shfl_down(s,  o, 32);
        s2 += __shfl_down(s2, o, 32);
    }
    s  = __shfl(s,  0, 32);
    s2 = __shfl(s2, 0, 32);
    const float mean = s / C;
    const float var  = s2 / C - mean * mean;
    const float inv  = rsqrtf(var + 1e-5f);
    float* yr = y + row * C;
    for (int i = threadIdx.x; i < C; i += 32)
        yr[i] = (xr[i] - mean) * inv * g[i] + b[i];
}

// ---------------------------------------------------------------------------
// roll(-ss) + window partition: (B,H,W,C) -> (B*nW, 64, C)
// ---------------------------------------------------------------------------
__global__ void winpart_kernel(const float* __restrict__ x, float* __restrict__ xw,
                               int Bb, int H, int W, int C, int ss, size_t n) {
    size_t i = (size_t)blockIdx.x * 256 + threadIdx.x;
    if (i >= n) return;
    int c = (int)(i % C);
    size_t t = i / C;
    int tw = (int)(t & 63);
    size_t wb = t >> 6;
    int nWx = W >> 3;
    int nW  = (H >> 3) * nWx;
    int b = (int)(wb / nW);
    int widx = (int)(wb % nW);
    int wh = widx / nWx, ww = widx % nWx;
    int y  = tw >> 3, xx = tw & 7;
    int hh = wh * 8 + y, wp = ww * 8 + xx;
    int sh = (hh + ss) % H, sw = (wp + ss) % W;
    xw[i] = x[(((size_t)b * H + sh) * W + sw) * C + c];
}

// window reverse + roll(+ss) + residual add: xo = shortcut + rev(wout)
__global__ void winrev_add_kernel(const float* __restrict__ wout,
                                  const float* __restrict__ shortcut,
                                  float* __restrict__ xo,
                                  int Bb, int H, int W, int C, int ss, size_t n) {
    size_t i = (size_t)blockIdx.x * 256 + threadIdx.x;
    if (i >= n) return;
    int c = (int)(i % C);
    size_t t = i / C;
    int w = (int)(t % W);
    size_t t2 = t / W;
    int hh = (int)(t2 % H);
    int b  = (int)(t2 / H);
    int sh = (hh - ss + H) % H, sw = (w - ss + W) % W;
    int nWx = W >> 3;
    int nW  = (H >> 3) * nWx;
    int win = b * nW + (sh >> 3) * nWx + (sw >> 3);
    int tw  = (sh & 7) * 8 + (sw & 7);
    xo[i] = shortcut[i] + wout[((size_t)win * 64 + tw) * C + c];
}

// ---------------------------------------------------------------------------
// channel-mod gating: bm = sigmoid(relu(relu(snr*w1+b1)@w2+b2)@w3+b3), 384 wide
// ---------------------------------------------------------------------------
__global__ void __launch_bounds__(384)
mod_gate_kernel(const float* __restrict__ snr,
                const float* __restrict__ w1, const float* __restrict__ b1,
                const float* __restrict__ w2, const float* __restrict__ b2,
                const float* __restrict__ w3, const float* __restrict__ b3,
                float* __restrict__ bm_out) {
    __shared__ float h1[384];
    __shared__ float h2[384];
    const int j = threadIdx.x;
    const float s = snr[0];
    h1[j] = fmaxf(0.f, s * w1[j] + b1[j]);
    __syncthreads();
    float a2 = b2[j];
    for (int i = 0; i < 384; ++i) a2 += h1[i] * w2[i * 384 + j];
    h2[j] = fmaxf(0.f, a2);
    __syncthreads();
    float a3 = b3[j];
    for (int i = 0; i < 384; ++i) a3 += h2[i] * w3[i * 384 + j];
    bm_out[j] = 1.0f / (1.0f + __expf(-a3));
}

__global__ void rowscale_kernel(float* __restrict__ t, const float* __restrict__ bm,
                                int C, size_t n) {
    size_t i = (size_t)blockIdx.x * 256 + threadIdx.x;
    if (i < n) t[i] *= bm[i % C];
}

__global__ void mul_kernel(float* __restrict__ x, const float* __restrict__ m, size_t n) {
    size_t i = (size_t)blockIdx.x * 256 + threadIdx.x;
    if (i < n) x[i] *= m[i];
}

// ---------------------------------------------------------------------------
// pixel shuffle: (B,H,W,4*od) -> (B,2H,2W,od); out[b,2h+i,2w+j,c] = y[b,h,w,c*4+i*2+j]
// ---------------------------------------------------------------------------
__global__ void pshuf_kernel(const float* __restrict__ y, float* __restrict__ o,
                             int Bb, int H, int W, int od, size_t n) {
    size_t i = (size_t)blockIdx.x * 256 + threadIdx.x;
    if (i >= n) return;
    int c = (int)(i % od);
    size_t t = i / od;
    int ow = (int)(t % (2 * W));
    size_t t2 = t / (2 * W);
    int oh = (int)(t2 % (2 * H));
    int b  = (int)(t2 / (2 * H));
    int hh = oh >> 1, ii = oh & 1;
    int wc = ow >> 1, jj = ow & 1;
    o[i] = y[(((size_t)b * H + hh) * W + wc) * (4 * od) + c * 4 + ii * 2 + jj];
}

// (B,HW,HW,3) -> (B,3,HW,HW)
__global__ void nchw_kernel(const float* __restrict__ x, float* __restrict__ o,
                            int Bb, int HW, size_t n) {
    size_t i = (size_t)blockIdx.x * 256 + threadIdx.x;
    if (i >= n) return;
    int w = (int)(i % HW);
    size_t t = i / HW;
    int hh = (int)(t % HW);
    size_t t2 = t / HW;
    int c = (int)(t2 % 3);
    int b = (int)(t2 / 3);
    o[i] = x[(((size_t)b * HW + hh) * HW + w) * 3 + c];
}

// ---------------------------------------------------------------------------
// Host orchestration
// ---------------------------------------------------------------------------
extern "C" void kernel_launch(void* const* d_in, const int* in_sizes, int n_in,
                              void* d_out, int out_size, void* d_ws, size_t ws_size,
                              hipStream_t stream) {
    (void)in_sizes; (void)n_in; (void)out_size; (void)ws_size;

    // ---- bind parameter leaves (jax pytree order: dict keys sorted) ----
    int ix = 0;
    auto nf = [&]() { return (const float*)d_in[ix++]; };
    const float* X0  = nf();   // x  (8, 1024, 32)
    const float* SNR = nf();   // snr (1,)
    const float* head_b = nf();
    const float* head_w = nf();
    struct BMp { const float *b1, *b2, *b3, *w1, *w2, *w3; } bmp[7];
    for (int i = 0; i < 7; ++i) {
        bmp[i].b1 = nf(); bmp[i].b2 = nf(); bmp[i].b3 = nf();
        bmp[i].w1 = nf(); bmp[i].w2 = nf(); bmp[i].w3 = nf();
    }
    const float* sm_b[8]; for (int i = 0; i < 8; ++i) sm_b[i] = nf();
    const float* sm_w[8]; for (int i = 0; i < 8; ++i) sm_w[i] = nf();
    struct Blk { const float *fc1_b,*fc1_w,*fc2_b,*fc2_w,*n1b,*n1g,*n2b,*n2g,
                             *proj_b,*proj_w,*qkv_b,*qkv_w,*rpb; };
    struct Upp { const float *inc_w, *nb, *ng; };
    Blk blk[3][2]; Upp up[3];
    for (int s = 0; s < 3; ++s) {
        for (int bi = 0; bi < 2; ++bi) {
            Blk& b = blk[s][bi];
            b.fc1_b = nf(); b.fc1_w = nf(); b.fc2_b = nf(); b.fc2_w = nf();
            b.n1b = nf(); b.n1g = nf(); b.n2b = nf(); b.n2g = nf();
            b.proj_b = nf(); b.proj_w = nf(); b.qkv_b = nf(); b.qkv_w = nf();
            b.rpb = nf();
        }
        up[s].inc_w = nf(); up[s].nb = nf(); up[s].ng = nf();
    }

    // ---- workspace carve ----
    float* ws = (float*)d_ws;
    size_t off = 0;
    auto carve = [&](size_t nelem) { float* p = ws + off; off += nelem; return p; };
    const size_t SZ_X   = (size_t)8 * 16384 * 64;    // max tokens*dim  (8.4M)
    const size_t SZ_QKV = (size_t)8 * 16384 * 192;   // max tokens*3dim
    const size_t SZ_HID = (size_t)8 * 16384 * 256;   // max tokens*4dim
    float* Xb   = carve(SZ_X);
    float* LNB  = carve(SZ_X);
    float* XW   = carve(SZ_X);
    float* AO   = carve(SZ_X);
    float* QKV  = carve(SZ_QKV);
    float* HID  = carve(SZ_HID);
    float* UPI  = carve(SZ_X);
    float* TPA  = carve((size_t)8192 * 384);
    float* TPB  = carve((size_t)8192 * 384);
    float* BMV  = carve(384);

    // ---- launch helpers ----
    auto gemm = [&](const float* A, const float* Wm, const float* bias,
                    const float* res, float* C, int M, int N, int K, int act) {
        dim3 g((unsigned)((N + 31) / 32), (unsigned)((M + 63) / 64));
        gemm_wmma_kernel<<<g, dim3(128), 0, stream>>>(A, Wm, bias, res, C, M, N, K, act);
    };
    auto ewgrid = [](size_t n) { return dim3((unsigned)((n + 255) / 256)); };

    const int EMBED[3] = {256, 128, 64};
    const int HEADS[3] = {8, 4, 2};
    const int Bb = 8;

    // ================= head: (8*1024,32) @ (32,256) + b =================
    gemm(X0, head_w, head_b, nullptr, Xb, Bb * 1024, 256, 32, 1);

    // ================= channel modulation (SA mode) =====================
    {
        const int M = Bb * 1024;
        const float* tin = Xb;
        float* tout = TPA;
        for (int i = 0; i < 7; ++i) {
            mod_gate_kernel<<<1, 384, 0, stream>>>(SNR, bmp[i].w1, bmp[i].b1,
                                                   bmp[i].w2, bmp[i].b2,
                                                   bmp[i].w3, bmp[i].b3, BMV);
            int K = (i == 0) ? 256 : 384;
            gemm(tin, sm_w[i], sm_b[i], nullptr, tout, M, 384, K, 1);
            size_t n = (size_t)M * 384;
            rowscale_kernel<<<ewgrid(n), 256, 0, stream>>>(tout, BMV, 384, n);
            tin = tout;
            tout = (tout == TPA) ? TPB : TPA;
        }
        // final: sigmoid(temp @ sm_w[7] + sm_b[7]) : 384 -> 256
        gemm(tin, sm_w[7], sm_b[7], nullptr, tout, M, 256, 384, 3);
        size_t n = (size_t)M * 256;
        mul_kernel<<<ewgrid(n), 256, 0, stream>>>(Xb, tout, n);
    }

    // ================= Swin stages ======================================
    int H = 32;
    for (int s = 0; s < 3; ++s) {
        const int dim = EMBED[s];
        const int nh  = HEADS[s];
        const int L   = H * H;
        const int tokens = Bb * L;
        const int nwb = Bb * (H / 8) * (H / 8);
        const size_t nelem = (size_t)tokens * dim;

        for (int bi = 0; bi < 2; ++bi) {
            const Blk& bp = blk[s][bi];
            const int ss = (bi == 0) ? 0 : 4;

            ln_kernel<<<dim3((unsigned)tokens), 32, 0, stream>>>(Xb, bp.n1g, bp.n1b, LNB, dim);
            winpart_kernel<<<ewgrid(nelem), 256, 0, stream>>>(LNB, XW, Bb, H, H, dim, ss, nelem);
            gemm(XW, bp.qkv_w, bp.qkv_b, nullptr, QKV, tokens, 3 * dim, dim, 1);
            attn_kernel<<<dim3((unsigned)nwb, (unsigned)nh), 256, 0, stream>>>(
                QKV, bp.rpb, AO, dim, nh, H, H, ss > 0 ? 1 : 0);
            gemm(AO, bp.proj_w, bp.proj_b, nullptr, XW, tokens, dim, dim, 1);
            winrev_add_kernel<<<ewgrid(nelem), 256, 0, stream>>>(XW, Xb, Xb, Bb, H, H, dim, ss, nelem);

            ln_kernel<<<dim3((unsigned)tokens), 32, 0, stream>>>(Xb, bp.n2g, bp.n2b, LNB, dim);
            gemm(LNB, bp.fc1_w, bp.fc1_b, nullptr, HID, tokens, 4 * dim, dim, 2);        // GELU
            gemm(HID, bp.fc2_w, bp.fc2_b, Xb, Xb, tokens, dim, 4 * dim, 1);              // +residual
        }

        // ---- upsample: LN -> Linear(dim, 4*od, no bias) -> pixel shuffle
        const int od = (s < 2) ? EMBED[s + 1] : 3;
        ln_kernel<<<dim3((unsigned)tokens), 32, 0, stream>>>(Xb, up[s].ng, up[s].nb, LNB, dim);
        gemm(LNB, up[s].inc_w, nullptr, nullptr, UPI, tokens, 4 * od, dim, 0);
        size_t nout = (size_t)Bb * 4 * H * H * od;
        pshuf_kernel<<<ewgrid(nout), 256, 0, stream>>>(UPI, Xb, Bb, H, H, od, nout);
        H *= 2;
    }

    // ================= final permute to (B,3,256,256) ===================
    {
        size_t n = (size_t)Bb * 3 * 256 * 256;
        nchw_kernel<<<ewgrid(n), 256, 0, stream>>>(Xb, (float*)d_out, Bb, 256, n);
    }
}
